// OurModel_71691594105502
// MI455X (gfx1250) — compile-verified, matrix operand order checked by reading, not run on hardware
//
#include <hip/hip_runtime.h>

// ---------------------------------------------------------------------------
// Model constants (from reference)
// ---------------------------------------------------------------------------
#define RANK      256
#define NPG       9
#define BGRAPH    8192
#define NNODES    (BGRAPH * NPG)     // 73728
#define NMAP      15625
#define MDD       2048

// GEMM tiling
#define ROWS      32                  // rows per block (2 M-tiles of 16)
#define HS        264                 // h_s row stride (bf16 elems): 528B, 16B-aligned rows
#define WS        40                  // w_s row stride (bf16 elems): 80B, 16B-aligned rows
#define YS        260                 // y_s row stride (f32 elems), padded
#define WBUF      (256 * WS)          // one W K-chunk buffer (bf16 elems)

#if __has_builtin(__builtin_amdgcn_global_load_async_to_lds_b128)
#define HAVE_ASYNC 1
#else
#define HAVE_ASYNC 0
#endif

#if __has_builtin(__builtin_amdgcn_s_wait_asynccnt)
#define WAIT_ASYNC(n) __builtin_amdgcn_s_wait_asynccnt(n)
#else
#define WAIT_ASYNC(n) asm volatile("s_wait_asynccnt %0" ::"n"(n) : "memory")
#endif

typedef __attribute__((ext_vector_type(16))) __bf16 v16bf;
typedef __attribute__((ext_vector_type(8)))  float  v8f;
typedef int v4i __attribute__((vector_size(16)));    // matches builtin param type

union Frag { v16bf v; unsigned int u[8]; };

__device__ __forceinline__ unsigned int f2bf1(float f) {
    unsigned int u = __float_as_uint(f);
    return (u + 0x7FFFu + ((u >> 16) & 1u)) >> 16;   // round-to-nearest-even
}
__device__ __forceinline__ unsigned int pack2(float lo, float hi) {
    return f2bf1(lo) | (f2bf1(hi) << 16);
}
__device__ __forceinline__ v8f wmma_bf16(const Frag& a, const Frag& b, v8f c) {
    return __builtin_amdgcn_wmma_f32_16x16x32_bf16(false, a.v, false, b.v,
                                                   (short)0, c, false, false);
}

// stage one W K-chunk row (64B) for thread t: global -> LDS
__device__ __forceinline__ void stage_chunk(const unsigned short* __restrict__ wch,
                                            unsigned short* __restrict__ wbuf,
                                            int kt, int t)
{
    const unsigned short* src = wch + (((size_t)kt << 8) + t) * 32;
    unsigned short* dst = wbuf + t * WS;
#if HAVE_ASYNC
    #pragma unroll
    for (int p = 0; p < 4; ++p)
        __builtin_amdgcn_global_load_async_to_lds_b128(
            (__attribute__((address_space(1))) v4i*)(src + p * 8),
            (__attribute__((address_space(3))) v4i*)(dst + p * 8), 0, 0);
#else
    #pragma unroll
    for (int p = 0; p < 4; ++p)
        *(uint4*)(dst + p * 8) = *(const uint4*)(src + p * 8);
#endif
}

// one K-chunk of the 32x256 x 256-chunk GEMM: build fragments, 4 WMMAs
__device__ __forceinline__ void frag_wmma(const unsigned short* __restrict__ h,
                                          const unsigned short* __restrict__ wbuf,
                                          int lane, int cw, int abase,
                                          v8f& c00, v8f& c01, v8f& c10, v8f& c11)
{
    const int half = lane >> 4, l15 = lane & 15;
    Frag a0, a1, b0, b1;
    const int ab = abase + half * 8;
    #pragma unroll
    for (int p = 0; p < 4; ++p) {
        a0.u[p]     = *(const unsigned int*)&h[l15 * HS + ab + 2 * p];
        a0.u[p + 4] = *(const unsigned int*)&h[l15 * HS + ab + 16 + 2 * p];
        a1.u[p]     = *(const unsigned int*)&h[(16 + l15) * HS + ab + 2 * p];
        a1.u[p + 4] = *(const unsigned int*)&h[(16 + l15) * HS + ab + 16 + 2 * p];
    }
    const int kb = half * 8;
    #pragma unroll
    for (int p = 0; p < 4; ++p) {
        b0.u[p]     = *(const unsigned int*)&wbuf[(cw + l15) * WS + kb + 2 * p];
        b0.u[p + 4] = *(const unsigned int*)&wbuf[(cw + l15) * WS + kb + 16 + 2 * p];
        b1.u[p]     = *(const unsigned int*)&wbuf[(cw + 16 + l15) * WS + kb + 2 * p];
        b1.u[p + 4] = *(const unsigned int*)&wbuf[(cw + 16 + l15) * WS + kb + 16 + 2 * p];
    }
    c00 = wmma_bf16(a0, b0, c00);
    c01 = wmma_bf16(a0, b1, c01);
    c10 = wmma_bf16(a1, b0, c10);
    c11 = wmma_bf16(a1, b1, c11);
}

// full K=256 loop, double-buffered async W staging when available
__device__ __forceinline__ void gemm_loop(const unsigned short* __restrict__ h_s,
                                          unsigned short* __restrict__ wbuf,  // [2][WBUF]
                                          const unsigned short* __restrict__ wch,
                                          int t, int lane, int cw,
                                          v8f& c00, v8f& c01, v8f& c10, v8f& c11)
{
#if HAVE_ASYNC
    stage_chunk(wch, wbuf, 0, t);
    stage_chunk(wch, wbuf + WBUF, 1, t);
    #pragma unroll
    for (int kt = 0; kt < 8; ++kt) {
        if (kt < 7) { WAIT_ASYNC(4); } else { WAIT_ASYNC(0); }
        __syncthreads();                    // also covers h_s stores on kt==0
        frag_wmma(h_s, wbuf + (kt & 1) * WBUF, lane, cw, kt * 32, c00, c01, c10, c11);
        __syncthreads();
        if (kt + 2 < 8) stage_chunk(wch, wbuf + (kt & 1) * WBUF, kt + 2, t);
    }
#else
    #pragma unroll
    for (int kt = 0; kt < 8; ++kt) {
        stage_chunk(wch, wbuf, kt, t);
        __syncthreads();
        frag_wmma(h_s, wbuf, lane, cw, kt * 32, c00, c01, c10, c11);
        __syncthreads();
    }
#endif
}

// epilogue: +bias into LDS, LayerNorm + ReLU, store rows to global
__device__ __forceinline__ void ln_epilogue(float* __restrict__ y_s,
                                            const v8f& c00, const v8f& c01,
                                            const v8f& c10, const v8f& c11,
                                            int t, int lane, int cw, int row0,
                                            const float* __restrict__ bias,
                                            const float* __restrict__ gam,
                                            const float* __restrict__ bet,
                                            float* __restrict__ y)
{
    {
        const int half = lane >> 4, l15 = lane & 15;
        const int col0 = cw + l15, col1 = cw + 16 + l15;
        const float bb0 = bias[col0], bb1 = bias[col1];
        #pragma unroll
        for (int i = 0; i < 8; ++i) {
            const int r = i + half * 8;
            y_s[r * YS + col0]        = c00[i] + bb0;
            y_s[r * YS + col1]        = c01[i] + bb1;
            y_s[(16 + r) * YS + col0] = c10[i] + bb0;
            y_s[(16 + r) * YS + col1] = c11[i] + bb1;
        }
    }
    __syncthreads();
    {
        const int r  = t >> 3;
        const int c0 = (t & 7) * 32;
        float v[32], s = 0.f, s2 = 0.f;
        #pragma unroll
        for (int i = 0; i < 32; ++i) {
            float u = y_s[r * YS + c0 + i];
            v[i] = u; s += u; s2 += u * u;
        }
        #pragma unroll
        for (int m = 4; m >= 1; m >>= 1) {
            s  += __shfl_xor(s,  m, 8);
            s2 += __shfl_xor(s2, m, 8);
        }
        const float mean = s * (1.f / 256.f);
        const float rstd = rsqrtf(s2 * (1.f / 256.f) - mean * mean + 1e-5f);
        float* yo = y + (size_t)(row0 + r) * RANK + c0;
        #pragma unroll
        for (int i = 0; i < 32; ++i) {
            const int c = c0 + i;
            float u = (v[i] - mean) * rstd * gam[c] + bet[c];
            yo[i] = fmaxf(u, 0.f);
        }
    }
}

// ---------------------------------------------------------------------------
// prep: convert weights to bf16 chunk-major [kt][n][32]; init inverse map
// ---------------------------------------------------------------------------
__global__ __launch_bounds__(256) void prep_kernel(
    const float* __restrict__ sage_W, const float* __restrict__ pred_W1,
    unsigned short* __restrict__ wsage, unsigned short* __restrict__ wpred,
    int* __restrict__ inv)
{
    const int idx = blockIdx.x * 256 + threadIdx.x;
    if (idx < 3 * 65536) {
        const int l    = idx >> 16;
        const int rem  = idx & 65535;
        const int kt   = rem >> 13;
        const int rem2 = rem & 8191;
        const int n    = rem2 >> 5;
        const int kk   = rem2 & 31;
        const int k    = kt * 32 + kk;
        wsage[idx] = (unsigned short)f2bf1(sage_W[((size_t)l << 16) + k * 256 + n]);
    } else if (idx < 4 * 65536) {
        const int rem  = idx - 3 * 65536;
        const int kt   = rem >> 13;
        const int rem2 = rem & 8191;
        const int n    = rem2 >> 5;
        const int kk   = rem2 & 31;
        const int k    = kt * 32 + kk;                 // first 256 rows of pred_W1
        wpred[rem] = (unsigned short)f2bf1(pred_W1[k * 256 + n]);
    } else if (idx < 4 * 65536 + NMAP) {
        inv[idx - 4 * 65536] = -1;
    }
}

// inv[key_idx[b]] = b  (key_idx unique by construction)
__global__ __launch_bounds__(256) void scatter_kernel(
    const int* __restrict__ key_ids, int* __restrict__ inv)
{
    const int b = blockIdx.x * 256 + threadIdx.x;
    if (b < BGRAPH) {
        const int* kp = key_ids + b * 6;
        int key = ((((kp[0] * 5 + kp[1]) * 5 + kp[2]) * 5 + kp[3]) * 5 + kp[4]) * 5 + kp[5];
        inv[key] = b;
    }
}

// ---------------------------------------------------------------------------
// SAGE layer 1 (fused embedding): x[j] = emb[features[j]]
// ---------------------------------------------------------------------------
__global__ __launch_bounds__(256) void sage_layer1_kernel(
    const int* __restrict__ features, const float* __restrict__ emb,
    float* __restrict__ y,
    const unsigned short* __restrict__ wch,
    const float* __restrict__ bias, const float* __restrict__ gam,
    const float* __restrict__ bet)
{
    __shared__ __align__(16) unsigned short h_s[ROWS * HS];
    __shared__ __align__(16) union { unsigned short w[2 * WBUF]; float y[ROWS * YS]; } wy;

    const int t    = threadIdx.x;
    const int lane = t & 31;
    const int w    = t >> 5;
    const int row0 = blockIdx.x * ROWS;

    {
        const int r  = t >> 3;
        const int c0 = (t & 7) * 32;
        const int j  = row0 + r;
        const int q  = j % NPG;
        const float inv = 1.0f / (1.0f + (q > 0 ? 1.0f : 0.0f) + (q < NPG - 1 ? 1.0f : 0.0f));
        const float* e0 = emb + features[j] * RANK + c0;
        const float* em = (q > 0)       ? emb + features[j - 1] * RANK + c0 : e0;
        const float* ep = (q < NPG - 1) ? emb + features[j + 1] * RANK + c0 : e0;
        #pragma unroll
        for (int cc = 0; cc < 32; cc += 4) {
            float4 s = *(const float4*)(e0 + cc);
            if (q > 0) {
                float4 p = *(const float4*)(em + cc);
                s.x += p.x; s.y += p.y; s.z += p.z; s.w += p.w;
            }
            if (q < NPG - 1) {
                float4 p = *(const float4*)(ep + cc);
                s.x += p.x; s.y += p.y; s.z += p.z; s.w += p.w;
            }
            *(unsigned int*)&h_s[r * HS + c0 + cc]     = pack2(s.x * inv, s.y * inv);
            *(unsigned int*)&h_s[r * HS + c0 + cc + 2] = pack2(s.z * inv, s.w * inv);
        }
    }

    v8f c00 = {0.f,0.f,0.f,0.f,0.f,0.f,0.f,0.f};
    v8f c01 = c00, c10 = c00, c11 = c00;
    const int cw = w * 32;
    gemm_loop(h_s, wy.w, wch, t, lane, cw, c00, c01, c10, c11);
    ln_epilogue(wy.y, c00, c01, c10, c11, t, lane, cw, row0, bias, gam, bet, y);
}

// ---------------------------------------------------------------------------
// SAGE layers 2..3: y = relu(LN( ((x[j-1]+x[j]+x[j+1])/deg) @ W + b ))
// ---------------------------------------------------------------------------
__global__ __launch_bounds__(256) void sage_layer_kernel(
    const float* __restrict__ x, float* __restrict__ y,
    const unsigned short* __restrict__ wch,
    const float* __restrict__ bias, const float* __restrict__ gam,
    const float* __restrict__ bet)
{
    __shared__ __align__(16) unsigned short h_s[ROWS * HS];
    __shared__ __align__(16) union { unsigned short w[2 * WBUF]; float y[ROWS * YS]; } wy;

    const int t    = threadIdx.x;
    const int lane = t & 31;
    const int w    = t >> 5;
    const int row0 = blockIdx.x * ROWS;

    {
        const int r  = t >> 3;
        const int c0 = (t & 7) * 32;
        const int j  = row0 + r;
        const int q  = j % NPG;
        const float inv = 1.0f / (1.0f + (q > 0 ? 1.0f : 0.0f) + (q < NPG - 1 ? 1.0f : 0.0f));
        const float* xj = x + (size_t)j * RANK + c0;
        #pragma unroll
        for (int cc = 0; cc < 32; cc += 4) {
            float4 s = *(const float4*)(xj + cc);
            if (q > 0) {
                float4 p = *(const float4*)(xj - RANK + cc);
                s.x += p.x; s.y += p.y; s.z += p.z; s.w += p.w;
            }
            if (q < NPG - 1) {
                float4 p = *(const float4*)(xj + RANK + cc);
                s.x += p.x; s.y += p.y; s.z += p.z; s.w += p.w;
            }
            *(unsigned int*)&h_s[r * HS + c0 + cc]     = pack2(s.x * inv, s.y * inv);
            *(unsigned int*)&h_s[r * HS + c0 + cc + 2] = pack2(s.z * inv, s.w * inv);
        }
    }

    v8f c00 = {0.f,0.f,0.f,0.f,0.f,0.f,0.f,0.f};
    v8f c01 = c00, c10 = c00, c11 = c00;
    const int cw = w * 32;
    gemm_loop(h_s, wy.w, wch, t, lane, cw, c00, c01, c10, c11);
    ln_epilogue(wy.y, c00, c01, c10, c11, t, lane, cw, row0, bias, gam, bet, y);
}

// ---------------------------------------------------------------------------
// dd path (single block): device-node row of the dnn-device SAGE + LN + ReLU,
// then info_contrib[c] = dd_out @ pred_W1[256:,:] + pred_b1
// ---------------------------------------------------------------------------
__global__ __launch_bounds__(256) void dd_kernel(
    const float* __restrict__ feats, const int* __restrict__ dd_src,
    const int* __restrict__ inv,
    const float* __restrict__ aug, const float* __restrict__ info_W,
    const float* __restrict__ info_b,
    const float* __restrict__ dd_W, const float* __restrict__ dd_b,
    const float* __restrict__ norm_g, const float* __restrict__ norm_b,
    const float* __restrict__ pred_W1, const float* __restrict__ pred_b1,
    float* __restrict__ info_contrib)
{
    __shared__ int   src_s[MDD];
    __shared__ float sh[256];
    __shared__ float red[256];
    __shared__ float mean_s, rstd_s;
    const int c = threadIdx.x;

    for (int j = c; j < MDD; j += 256) src_s[j] = dd_src[j];
    __syncthreads();

    float info = info_b[c];
    #pragma unroll
    for (int k = 0; k < 5; ++k) info += aug[k] * info_W[k * 256 + c];

    float acc = 0.f;
    for (int j = 0; j < MDD; ++j) {
        const int b = inv[src_s[j]];
        if (b >= 0) acc += feats[(size_t)b * (NPG * RANK) + c];
    }
    const float h = (acc + info) * (1.0f / 2049.0f);
    sh[c] = h;
    __syncthreads();

    float o = dd_b[c];
    for (int k = 0; k < 256; ++k) o += sh[k] * dd_W[k * 256 + c];

    red[c] = o; __syncthreads();
    for (int s = 128; s > 0; s >>= 1) { if (c < s) red[c] += red[c + s]; __syncthreads(); }
    if (c == 0) mean_s = red[0] * (1.f / 256.f);
    __syncthreads();
    const float d = o - mean_s;
    red[c] = d * d; __syncthreads();
    for (int s = 128; s > 0; s >>= 1) { if (c < s) red[c] += red[c + s]; __syncthreads(); }
    if (c == 0) rstd_s = rsqrtf(red[0] * (1.f / 256.f) + 1e-5f);
    __syncthreads();
    float v = d * rstd_s * norm_g[c] + norm_b[c];
    v = fmaxf(v, 0.f);
    __syncthreads();
    sh[c] = v;
    __syncthreads();

    float ic = pred_b1[c];
    for (int k = 0; k < 256; ++k) ic += sh[k] * pred_W1[(256 + k) * 256 + c];
    info_contrib[c] = ic;
}

// ---------------------------------------------------------------------------
// predictor: out[g] = relu(graph_emb[g] @ W1a + info_contrib) @ W2 + b2
// ---------------------------------------------------------------------------
__global__ __launch_bounds__(256) void pred_kernel(
    const float* __restrict__ feats,              // [NNODES][256], graph row = g*9
    const unsigned short* __restrict__ wch,       // bf16 [8][256][32] of W1[:256,:]^T
    const float* __restrict__ icv,                // [256]
    const float* __restrict__ W2, const float* __restrict__ b2,
    float* __restrict__ out)
{
    __shared__ __align__(16) unsigned short h_s[ROWS * HS];
    __shared__ __align__(16) union { unsigned short w[2 * WBUF]; float y[ROWS * YS]; } wy;

    const int t    = threadIdx.x;
    const int lane = t & 31;
    const int w    = t >> 5;
    const int row0 = blockIdx.x * ROWS;

    {
        const int r  = t >> 3;
        const int c0 = (t & 7) * 32;
        const float* xg = feats + (size_t)(row0 + r) * (NPG * RANK) + c0;
        #pragma unroll
        for (int cc = 0; cc < 32; cc += 4) {
            float4 s = *(const float4*)(xg + cc);
            *(unsigned int*)&h_s[r * HS + c0 + cc]     = pack2(s.x, s.y);
            *(unsigned int*)&h_s[r * HS + c0 + cc + 2] = pack2(s.z, s.w);
        }
    }

    v8f c00 = {0.f,0.f,0.f,0.f,0.f,0.f,0.f,0.f};
    v8f c01 = c00, c10 = c00, c11 = c00;
    const int cw = w * 32;
    gemm_loop(h_s, wy.w, wch, t, lane, cw, c00, c01, c10, c11);

    float* y_s = wy.y;
    {
        const int half = lane >> 4, l15 = lane & 15;
        const int col0 = cw + l15, col1 = cw + 16 + l15;
        const float ic0 = icv[col0], ic1 = icv[col1];
        const float w20 = W2[col0],  w21 = W2[col1];
        #pragma unroll
        for (int i = 0; i < 8; ++i) {
            const int r = i + half * 8;
            y_s[r * YS + col0]        = fmaxf(c00[i] + ic0, 0.f) * w20;
            y_s[r * YS + col1]        = fmaxf(c01[i] + ic1, 0.f) * w21;
            y_s[(16 + r) * YS + col0] = fmaxf(c10[i] + ic0, 0.f) * w20;
            y_s[(16 + r) * YS + col1] = fmaxf(c11[i] + ic1, 0.f) * w21;
        }
    }
    __syncthreads();
    {
        const int r  = t >> 3;
        const int c0 = (t & 7) * 32;
        float s = 0.f;
        #pragma unroll
        for (int i = 0; i < 32; ++i) s += y_s[r * YS + c0 + i];
        #pragma unroll
        for (int m = 4; m >= 1; m >>= 1) s += __shfl_xor(s, m, 8);
        if ((t & 7) == 0) out[row0 + r] = s + b2[0];
    }
}

// ---------------------------------------------------------------------------
// launch
// ---------------------------------------------------------------------------
extern "C" void kernel_launch(void* const* d_in, const int* in_sizes, int n_in,
                              void* d_out, int out_size, void* d_ws, size_t ws_size,
                              hipStream_t stream)
{
    (void)in_sizes; (void)n_in; (void)out_size; (void)ws_size;

    const int*   features = (const int*)d_in[0];
    const int*   key_ids  = (const int*)d_in[1];
    const int*   dd_src   = (const int*)d_in[4];
    const float* emb      = (const float*)d_in[5];
    const float* sage_W   = (const float*)d_in[6];
    const float* sage_b   = (const float*)d_in[7];
    const float* ln_g     = (const float*)d_in[8];
    const float* ln_b     = (const float*)d_in[9];
    const float* dd_W     = (const float*)d_in[10];
    const float* dd_b     = (const float*)d_in[11];
    const float* norm_g   = (const float*)d_in[12];
    const float* norm_b   = (const float*)d_in[13];
    const float* info_W   = (const float*)d_in[14];
    const float* info_b   = (const float*)d_in[15];
    const float* aug      = (const float*)d_in[16];
    const float* pred_W1  = (const float*)d_in[17];
    const float* pred_b1  = (const float*)d_in[18];
    const float* pred_W2  = (const float*)d_in[19];
    const float* pred_b2  = (const float*)d_in[20];
    float* out = (float*)d_out;

    char* ws = (char*)d_ws;
    const size_t NB = (size_t)NNODES * RANK * sizeof(float);   // 75,497,472
    float*          featsA = (float*)(ws);
    float*          featsB = (float*)(ws + NB);
    unsigned short* wsage  = (unsigned short*)(ws + 2 * NB);
    unsigned short* wpred  = (unsigned short*)(ws + 2 * NB + 3 * 65536 * 2);
    int*            inv    = (int*)(ws + 2 * NB + 4 * 65536 * 2);
    float*          icv    = (float*)(ws + 2 * NB + 4 * 65536 * 2 + 62720);

    prep_kernel   <<<1086, 256, 0, stream>>>(sage_W, pred_W1, wsage, wpred, inv);
    scatter_kernel<<<  32, 256, 0, stream>>>(key_ids, inv);

    // layer 1 (fused embedding gather) -> featsA
    sage_layer1_kernel<<<NNODES / ROWS, 256, 0, stream>>>(features, emb, featsA, wsage,
                                                          sage_b, ln_g, ln_b);
    // layer 2 -> featsB
    sage_layer_kernel<<<NNODES / ROWS, 256, 0, stream>>>(featsA, featsB, wsage + 65536,
                                                         sage_b + 256, ln_g + 256, ln_b + 256);
    // layer 3 -> featsA
    sage_layer_kernel<<<NNODES / ROWS, 256, 0, stream>>>(featsB, featsA, wsage + 131072,
                                                         sage_b + 512, ln_g + 512, ln_b + 512);

    dd_kernel<<<1, 256, 0, stream>>>(featsA, dd_src, inv, aug, info_W, info_b,
                                     dd_W, dd_b, norm_g, norm_b,
                                     pred_W1, pred_b1, icv);

    pred_kernel<<<BGRAPH / ROWS, 256, 0, stream>>>(featsA, wpred, icv,
                                                   pred_W2, pred_b2, out);
}
